// MoeMLP_58703613002486
// MI455X (gfx1250) — matrix operand order repaired and checked by reading, not compiled, not run on hardware
//
#include <hip/hip_runtime.h>
#include <math.h>

// ---------------- constants (match reference) ----------------
#define NUM_EXPERTS 8
#define TOP_K       2
#define N_EMBD      1024
#define D_FFN       2048
#define BLOCK_M     128
#define N_TOKENS    4096            // 2*2048
#define M_ROWS      (N_TOKENS*TOP_K) // 8192
#define N_BLOCKS    (M_ROWS/BLOCK_M) // 64
#define W1_COLS     (NUM_EXPERTS*D_FFN) // 16384
#define OUT_GEMM_ELEMS ((size_t)M_ROWS*(size_t)D_FFN) // 16777216

typedef __attribute__((ext_vector_type(16))) _Float16 v16h;
typedef __attribute__((ext_vector_type(8)))  float    v8f;
typedef __attribute__((ext_vector_type(4)))  float    v4f;
typedef __attribute__((ext_vector_type(4)))  unsigned int v4u;

union FragU { v4u u[2]; v16h h; _Float16 f[16]; };

__device__ __forceinline__ float gelu_exact(float v) {
    return 0.5f * v * (1.0f + erff(v * 0.70710678118654752f));
}

// ============================================================
// Kernel 1: router logits (f32 exact), softmax, top-2, weights
// grid: 32 blocks x 128 threads; one thread per token
// ============================================================
__global__ void __launch_bounds__(128)
moe_router_topk(const float* __restrict__ x,
                const float* __restrict__ router_w,
                float* __restrict__ logits_out,   // [4096, 8]
                int*   __restrict__ sel_rep,      // [8192]
                float* __restrict__ rw_rep)       // [8192]
{
    __shared__ float s_rw[NUM_EXPERTS * N_EMBD]; // 32KB
    const int tid = threadIdx.x;
    for (int i = tid; i < NUM_EXPERTS * N_EMBD; i += 128)
        s_rw[i] = router_w[i];
    __syncthreads();

    const int tok = blockIdx.x * 128 + tid;
    const float* xr = x + (size_t)tok * N_EMBD;

    float acc[NUM_EXPERTS];
#pragma unroll
    for (int e = 0; e < NUM_EXPERTS; ++e) acc[e] = 0.0f;

    for (int k = 0; k < N_EMBD; k += 4) {
        v4f xv = *(const v4f*)(xr + k);
#pragma unroll
        for (int e = 0; e < NUM_EXPERTS; ++e) {
            v4f wv = *(const v4f*)(&s_rw[e * N_EMBD + k]); // broadcast LDS read
            acc[e] += xv.x * wv.x + xv.y * wv.y + xv.z * wv.z + xv.w * wv.w;
        }
    }

#pragma unroll
    for (int e = 0; e < NUM_EXPERTS; ++e)
        logits_out[tok * NUM_EXPERTS + e] = acc[e];

    // softmax (f32)
    float mx = acc[0];
#pragma unroll
    for (int e = 1; e < NUM_EXPERTS; ++e) mx = fmaxf(mx, acc[e]);
    float p[NUM_EXPERTS], psum = 0.0f;
#pragma unroll
    for (int e = 0; e < NUM_EXPERTS; ++e) { p[e] = __expf(acc[e] - mx); psum += p[e]; }
    float inv = 1.0f / psum;
#pragma unroll
    for (int e = 0; e < NUM_EXPERTS; ++e) p[e] *= inv;

    // top-2: strict '>' scans -> first-index tie-break (matches lax.top_k)
    int a = 0;
#pragma unroll
    for (int e = 1; e < NUM_EXPERTS; ++e) if (acc[e] > acc[a]) a = e;
    int b = (a == 0) ? 1 : 0;
#pragma unroll
    for (int e = 0; e < NUM_EXPERTS; ++e) {
        if (e == a) continue;
        if (acc[e] > acc[b]) b = e;
    }
    float wsum = p[a] + p[b];
    float wa = p[a] / wsum, wb = p[b] / wsum;

    sel_rep[2 * tok + 0] = a;  rw_rep[2 * tok + 0] = wa;
    sel_rep[2 * tok + 1] = b;  rw_rep[2 * tok + 1] = wb;
}

// ============================================================
// Kernel 2: stable counting sort by expert (deterministic)
// single workgroup, 256 threads (8 do the sequential emit)
// ============================================================
__global__ void __launch_bounds__(256)
moe_group_sort(const int*   __restrict__ sel_rep,
               const float* __restrict__ rw_rep,
               int*   __restrict__ tok_sorted,   // [8192] source token per dest row
               float* __restrict__ rw_sorted,    // [8192]
               int*   __restrict__ block_expert) // [64]
{
    __shared__ int s_sel[M_ROWS];   // 32KB
    __shared__ int s_cnt[NUM_EXPERTS];
    __shared__ int s_off[NUM_EXPERTS];
    const int tid = threadIdx.x;

    for (int i = tid; i < M_ROWS; i += 256) s_sel[i] = sel_rep[i];
    __syncthreads();

    if (tid < NUM_EXPERTS) {
        int c = 0;
        for (int i = 0; i < M_ROWS; ++i) c += (s_sel[i] == tid) ? 1 : 0;
        s_cnt[tid] = c;
    }
    __syncthreads();
    if (tid == 0) {
        int o = 0;
        for (int e = 0; e < NUM_EXPERTS; ++e) { s_off[e] = o; o += s_cnt[e]; }
    }
    __syncthreads();
    if (tid < NUM_EXPERTS) {
        int pos = s_off[tid];
        for (int i = 0; i < M_ROWS; ++i) {
            if (s_sel[i] == tid) {
                tok_sorted[pos] = i >> 1;          // x_rep row i -> token i/2
                rw_sorted[pos]  = rw_rep[i];
                if ((pos & (BLOCK_M - 1)) == 0) block_expert[pos >> 7] = tid;
                ++pos;
            }
        }
    }
}

// ============================================================
// Kernel 3: block-sparse GEMM via v_wmma_f32_16x16x32_f16
// grid: (16 col-tiles, 64 row-blocks) x 256 threads (8 waves)
// Each WG: 128x128 output tile; wave w owns rows [16w,16w+16)
// ============================================================
#define LDS_STRIDE 40   // 32 halves + 8 pad -> conflict-free b128 frag loads

__global__ void __launch_bounds__(256)
moe_block_gemm(const float* __restrict__ x,
               const float* __restrict__ w1,
               const int*   __restrict__ tok_sorted,
               const float* __restrict__ rw_sorted,
               const int*   __restrict__ block_expert,
               float* __restrict__ out)           // [8192, 2048]
{
    __shared__ _Float16 sA[BLOCK_M * LDS_STRIDE]; // 10KB  [m][k]
    __shared__ _Float16 sB[128 * LDS_STRIDE];     // 10KB  [n][k] (transposed)
    __shared__ int   sTok[BLOCK_M];
    __shared__ float sRw[BLOCK_M];

    const int tid  = threadIdx.x;
    const int blk  = blockIdx.y;            // 0..63 row block
    const int cb   = blockIdx.x * 128;      // column base
    const int e    = block_expert[blk];     // uniform scalar load

    if (tid < BLOCK_M) {
        sTok[tid] = tok_sorted[blk * BLOCK_M + tid];
        sRw[tid]  = rw_sorted[blk * BLOCK_M + tid];
    }
    __syncthreads();

    const int wave = tid >> 5;
    const int lane = tid & 31;
    const int m15  = lane & 15;
    const int hi   = lane >> 4;

    v8f acc[8];
#pragma unroll
    for (int ct = 0; ct < 8; ++ct) acc[ct] = (v8f){0,0,0,0,0,0,0,0};

    // staging indices (A): thread -> (row, 16-float half of the K-chunk)
    const int a_row  = tid >> 1;
    const int a_half = tid & 1;
    const int a_tok  = sTok[a_row];

    for (int k0 = 0; k0 < N_EMBD; k0 += 32) {
        // ---- stage A: 128 rows x 32 K, gathered, f32 -> f16
        {
            const float* src = x + (size_t)a_tok * N_EMBD + k0 + a_half * 16;
            v4f f0 = *(const v4f*)(src + 0);
            v4f f1 = *(const v4f*)(src + 4);
            v4f f2 = *(const v4f*)(src + 8);
            v4f f3 = *(const v4f*)(src + 12);
            FragU pk;
            pk.f[0]=(_Float16)f0.x; pk.f[1]=(_Float16)f0.y; pk.f[2]=(_Float16)f0.z; pk.f[3]=(_Float16)f0.w;
            pk.f[4]=(_Float16)f1.x; pk.f[5]=(_Float16)f1.y; pk.f[6]=(_Float16)f1.z; pk.f[7]=(_Float16)f1.w;
            pk.f[8]=(_Float16)f2.x; pk.f[9]=(_Float16)f2.y; pk.f[10]=(_Float16)f2.z; pk.f[11]=(_Float16)f2.w;
            pk.f[12]=(_Float16)f3.x; pk.f[13]=(_Float16)f3.y; pk.f[14]=(_Float16)f3.z; pk.f[15]=(_Float16)f3.w;
            _Float16* dst = &sA[a_row * LDS_STRIDE + a_half * 16];
            *(v4u*)(dst + 0) = pk.u[0];
            *(v4u*)(dst + 8) = pk.u[1];
        }
        // ---- stage B: 32 K x 128 N of expert e, transposed to [n][k], f32 -> f16
#pragma unroll
        for (int i = 0; i < 4; ++i) {
            int f4 = tid + i * 256;           // 0..1023 float4s
            int kk = f4 >> 5;                 // 0..31
            int n4 = (f4 & 31) * 4;           // 0..124
            const float* wsrc = w1 + (size_t)(k0 + kk) * W1_COLS + e * D_FFN + cb + n4;
            v4f wv = *(const v4f*)wsrc;
            sB[(n4 + 0) * LDS_STRIDE + kk] = (_Float16)wv.x;
            sB[(n4 + 1) * LDS_STRIDE + kk] = (_Float16)wv.y;
            sB[(n4 + 2) * LDS_STRIDE + kk] = (_Float16)wv.z;
            sB[(n4 + 3) * LDS_STRIDE + kk] = (_Float16)wv.w;
        }
        __syncthreads();

        // ---- compute: A frag per ISA layout (lanes0-15: K=[hi*8..)+[16+hi*8..))
        FragU a;
        const _Float16* arow = &sA[(wave * 16 + m15) * LDS_STRIDE];
        a.u[0] = *(const v4u*)(arow + hi * 8);
        a.u[1] = *(const v4u*)(arow + 16 + hi * 8);
#pragma unroll
        for (int ct = 0; ct < 8; ++ct) {
            FragU bf;
            const _Float16* brow = &sB[(ct * 16 + m15) * LDS_STRIDE];
            bf.u[0] = *(const v4u*)(brow + hi * 8);
            bf.u[1] = *(const v4u*)(brow + 16 + hi * 8);
            acc[ct] = __builtin_amdgcn_wmma_f32_16x16x32_f16(
                false, a.h, false, bf.h, (short)0, acc[ct], false, false);
        }
        __syncthreads();
    }

    // ---- epilogue: C/D layout VGPR r -> M = r + 8*hi, N = lane&15
    const int rowLoc = wave * 16 + hi * 8;
    const size_t rowG = (size_t)blk * BLOCK_M + rowLoc;
#pragma unroll
    for (int ct = 0; ct < 8; ++ct) {
        const int colG = cb + ct * 16 + m15;
#pragma unroll
        for (int r = 0; r < 8; ++r) {
            float v  = acc[ct][r];
            float rw = sRw[rowLoc + r];
            out[(rowG + r) * (size_t)D_FFN + colG] = gelu_exact(v) * rw;
        }
    }
}

// ============================================================
extern "C" void kernel_launch(void* const* d_in, const int* in_sizes, int n_in,
                              void* d_out, int out_size, void* d_ws, size_t ws_size,
                              hipStream_t stream) {
    const float* x        = (const float*)d_in[0];   // [2,2048,1024]
    const float* router_w = (const float*)d_in[1];   // [8,1024]
    const float* w1       = (const float*)d_in[2];   // [1024, 16384]

    float* out_gemm   = (float*)d_out;                          // [8192,2048]
    float* out_logits = (float*)d_out + OUT_GEMM_ELEMS;         // [4096,8]

    int*   sel_rep      = (int*)d_ws;                 // 8192 ints
    float* rw_rep       = (float*)d_ws + 8192;        // 8192 floats
    int*   tok_sorted   = (int*)d_ws + 16384;         // 8192 ints
    float* rw_sorted    = (float*)d_ws + 24576;       // 8192 floats
    int*   block_expert = (int*)d_ws + 32768;         // 64 ints

    moe_router_topk<<<dim3(N_TOKENS / 128), dim3(128), 0, stream>>>(
        x, router_w, out_logits, sel_rep, rw_rep);

    moe_group_sort<<<dim3(1), dim3(256), 0, stream>>>(
        sel_rep, rw_rep, tok_sorted, rw_sorted, block_expert);

    moe_block_gemm<<<dim3(D_FFN / 128, N_BLOCKS), dim3(256), 0, stream>>>(
        x, w1, tok_sorted, rw_sorted, block_expert, out_gemm);
}
